// GRNNet_25125558682314
// MI455X (gfx1250) — matrix operand order repaired and checked by reading, not compiled, not run on hardware
//
#include <hip/hip_runtime.h>

typedef __attribute__((ext_vector_type(16))) _Float16 v16h;
typedef __attribute__((ext_vector_type(8)))  _Float16 v8h;
typedef __attribute__((ext_vector_type(8)))  float    v8f;
typedef __attribute__((ext_vector_type(2)))  float    v2f;

#define N_TRAIN   16384
#define DIM       64
#define OUT_DIM   8
#define BATCH     8192
#define S_SLICES  8
#define NPER      (N_TRAIN / S_SLICES)   // 2048 training pts per slice
#define TILES_PER (NPER / 16)            // 128 tiles per wave-unit
#define ROWTILES  (BATCH / 16)           // 512
#define WPB       4                      // waves per block

// fast hardware transcendentals (v_sqrt_f32 / v_exp_f32), ~1ulp — plenty for
// exp(-beta*d): a 1e-6 error in d perturbs weights by exp(1e-5).
__device__ __forceinline__ float fast_sqrt(float x) {
#if __has_builtin(__builtin_amdgcn_sqrtf)
  return __builtin_amdgcn_sqrtf(x);
#else
  return sqrtf(x);
#endif
}
__device__ __forceinline__ float fast_exp2(float x) {
#if __has_builtin(__builtin_amdgcn_exp2f)
  return __builtin_amdgcn_exp2f(x);
#else
  return exp2f(x);
#endif
}

// ---------------- pre-pass kernels ----------------

// f32 -> f16 hi/lo split (Ozaki-style): v = hi + lo with ~22-bit mantissa coverage
__global__ void split_f32(const float* __restrict__ src, _Float16* __restrict__ hi,
                          _Float16* __restrict__ lo, int count) {
  int i = blockIdx.x * blockDim.x + threadIdx.x;
  if (i < count) {
    float v = src[i];
    _Float16 h = (_Float16)v;
    hi[i] = h;
    lo[i] = (_Float16)(v - (float)h);
  }
}

// one wave per row: ||row||^2 in fp32
__global__ void row_norm(const float* __restrict__ src, float* __restrict__ out) {
  int row = blockIdx.x;
  int lane = threadIdx.x;               // block = 32 (one wave32)
  const float* p = src + row * DIM + lane * 2;
  float a = p[0], b = p[1];
  float t = a * a + b * b;
  for (int off = 16; off > 0; off >>= 1) t += __shfl_xor(t, off, 32);
  if (lane == 0) out[row] = t;
}

// Yp[n][0..7]=Y[n][:], Yp[n][8]=1 (psum1 column), Yp[n][9..15]=0
__global__ void pad_y(const float* __restrict__ y, float* __restrict__ yp) {
  int i = blockIdx.x * blockDim.x + threadIdx.x;   // N_TRAIN*16
  int j = i & 15;
  int n = i >> 4;
  yp[i] = (j < OUT_DIM) ? y[n * OUT_DIM + j] : (j == OUT_DIM ? 1.0f : 0.0f);
}

// ---------------- fused main kernel ----------------
// Each wave: 16 batch rows x 2048 training points (one N-slice), streaming
// 16x16 tiles.  Stage1: S = A @ X^T via 6x f16 WMMA (hi/lo split, fp32 acc).
// Epilogue: w = exp(-beta*sqrt(max(b2+x2-2S,0))) in registers (3 ops/elem).
// Stage2: O += W @ [Y|1] via f32 WMMA after an in-wave LDS transpose of W.
__global__ __launch_bounds__(WPB * 32) void grnn_main(
    const _Float16* __restrict__ bhi, const _Float16* __restrict__ blo,
    const _Float16* __restrict__ xhi, const _Float16* __restrict__ xlo,
    const float* __restrict__ b2,  const float* __restrict__ x2,
    const float* __restrict__ beta, const float* __restrict__ yp,
    float* __restrict__ part)
{
  __shared__ float lds[WPB * 256];      // 16x16 f32 W tile per wave
  const int lane = threadIdx.x & 31;
  const int wave = threadIdx.x >> 5;
  const int unit = blockIdx.x * WPB + wave;     // 0..4095
  const int r = unit / S_SLICES;                // batch row-tile
  const int s = unit % S_SLICES;                // N slice
  const int half = lane >> 4;                   // wave half (K split)
  const int col  = lane & 15;
  float* wl = lds + wave * 256;

  // A operands (16 batch rows, K=64 as two K=32 chunks), resident in VGPRs.
  // f16 A k order used here (same as B): lane holds row M=lane%16; halves 0..15 -> K=K0+16*half+i.
  const int arow = r * 16 + col;
  v16h ahi[2], alo[2];
  #pragma unroll
  for (int c = 0; c < 2; ++c) {
    const int base = arow * DIM + 32 * c + 16 * half;
    v8h h0 = *(const v8h*)(bhi + base);
    v8h h1 = *(const v8h*)(bhi + base + 8);
    v8h l0 = *(const v8h*)(blo + base);
    v8h l1 = *(const v8h*)(blo + base + 8);
    #pragma unroll
    for (int i = 0; i < 8; ++i) {
      ahi[c][i] = h0[i]; ahi[c][i + 8] = h1[i];
      alo[c][i] = l0[i]; alo[c][i + 8] = l1[i];
    }
  }
  // C/D layout rows for this lane: M = v + 8*half
  v8f b2v  = *(const v8f*)(b2 + r * 16 + 8 * half);
  v8f accO = {};                                // persistent [Y|1] accumulator

  for (int t = 0; t < TILES_PER; ++t) {
    const int tb = s * NPER + t * 16;
    const int n  = tb + col;                    // this lane's training column
    // f16 B k order used here, same as A (32x16): lanes 0-15 K=0..15, lanes 16-31 K=16..31 -> 16
    // contiguous halves per lane.
    const int xb = n * DIM + 16 * half;
    v16h bh0 = *(const v16h*)(xhi + xb);
    v16h bh1 = *(const v16h*)(xhi + xb + 32);
    v16h bl0 = *(const v16h*)(xlo + xb);
    v16h bl1 = *(const v16h*)(xlo + xb + 32);
    const float x2n = x2[n];
    const float nb  = -beta[n] * 1.44269504088896340736f;  // -beta*log2(e)

    // S = Ah*Bh + Ah*Bl + Al*Bh  (fp32 accumulate)
    v8f accS = {};
    accS = __builtin_amdgcn_wmma_f32_16x16x32_f16(false, ahi[0], false, bh0, (short)0, accS, false, false);
    accS = __builtin_amdgcn_wmma_f32_16x16x32_f16(false, ahi[1], false, bh1, (short)0, accS, false, false);
    accS = __builtin_amdgcn_wmma_f32_16x16x32_f16(false, ahi[0], false, bl0, (short)0, accS, false, false);
    accS = __builtin_amdgcn_wmma_f32_16x16x32_f16(false, ahi[1], false, bl1, (short)0, accS, false, false);
    accS = __builtin_amdgcn_wmma_f32_16x16x32_f16(false, alo[0], false, bh0, (short)0, accS, false, false);
    accS = __builtin_amdgcn_wmma_f32_16x16x32_f16(false, alo[1], false, bh1, (short)0, accS, false, false);

    // RBF epilogue in registers; scatter W tile to LDS as [m][n]
    #pragma unroll
    for (int v = 0; v < 8; ++v) {
      float sq = fmaf(-2.0f, accS[v], b2v[v] + x2n);
      sq = fmaxf(sq, 0.0f);
      float w = fast_exp2(nb * fast_sqrt(sq));
      wl[(v + 8 * half) * 16 + col] = w;
    }

    // O += W(16x16) @ Yp(16x16), K in chunks of 4 via f32 WMMA.
    // A f32 layout: lane row M=lane%16, VGPR v -> K = k0 + 2*half + v
    // B f32 layout: lane col N=lane%16, VGPR v -> K = k0 + 2*half + v
    #pragma unroll
    for (int k = 0; k < 4; ++k) {
      const int k0 = 4 * k;
      v2f wa = *(const v2f*)(wl + col * 16 + k0 + 2 * half);
      v2f yb;
      yb[0] = yp[(tb + k0 + 2 * half)     * 16 + col];
      yb[1] = yp[(tb + k0 + 2 * half + 1) * 16 + col];
      accO = __builtin_amdgcn_wmma_f32_16x16x4_f32(false, wa, false, yb, (short)0, accO, false, false);
    }
  }

  // partial tile for (row-tile r, slice s): [16 rows][16 cols] (cols 0..8 used)
  float* dst = part + ((r * S_SLICES + s) * 16) * 16;
  #pragma unroll
  for (int v = 0; v < 8; ++v)
    dst[(v + 8 * half) * 16 + col] = accO[v];
}

// ---------------- deterministic reduction + divide ----------------
__global__ void grnn_reduce(const float* __restrict__ part, float* __restrict__ out) {
  int tid = blockIdx.x * blockDim.x + threadIdx.x;   // BATCH*OUT_DIM
  int m = tid >> 3, j = tid & 7;
  int r = m >> 4, m16 = m & 15;
  float s1 = 0.0f, s2 = 0.0f;
  for (int s = 0; s < S_SLICES; ++s) {
    const float* p = part + ((r * S_SLICES + s) * 16 + m16) * 16;
    s2 += p[j];
    s1 += p[OUT_DIM];       // ones-column = psum1
  }
  out[tid] = s2 / s1;
}

// ---------------- host launcher ----------------
extern "C" void kernel_launch(void* const* d_in, const int* in_sizes, int n_in,
                              void* d_out, int out_size, void* d_ws, size_t ws_size,
                              hipStream_t stream) {
  const float* batches = (const float*)d_in[0];   // [8192, 64]
  const float* x_train = (const float*)d_in[1];   // [16384, 64]
  const float* y_train = (const float*)d_in[2];   // [16384, 8]
  const float* beta    = (const float*)d_in[3];   // [1, 16384]

  char* ws = (char*)d_ws;
  _Float16* xhi = (_Float16*)(ws);                          // 2 MB
  _Float16* xlo = (_Float16*)(ws + (2u << 20));             // 2 MB
  _Float16* bhi = (_Float16*)(ws + (4u << 20));             // 1 MB
  _Float16* blo = (_Float16*)(ws + (5u << 20));             // 1 MB
  float*    x2  = (float*)(ws + (6u << 20));                // 64 KB
  float*    b2  = (float*)(ws + (6u << 20) + (128u << 10)); // 32 KB
  float*    yp  = (float*)(ws + (6u << 20) + (256u << 10)); // 1 MB
  float*    part= (float*)(ws + (8u << 20));                // 4 MB

  // pre-pass: hi/lo splits, row norms, padded Y
  split_f32<<<(N_TRAIN * DIM + 255) / 256, 256, 0, stream>>>(x_train, xhi, xlo, N_TRAIN * DIM);
  split_f32<<<(BATCH  * DIM + 255) / 256, 256, 0, stream>>>(batches, bhi, blo, BATCH * DIM);
  row_norm<<<N_TRAIN, 32, 0, stream>>>(x_train, x2);
  row_norm<<<BATCH,   32, 0, stream>>>(batches, b2);
  pad_y<<<(N_TRAIN * 16 + 255) / 256, 256, 0, stream>>>(y_train, yp);

  // fused GEMM + RBF + contraction: 512 row-tiles x 8 slices = 4096 waves
  grnn_main<<<ROWTILES * S_SLICES / WPB, WPB * 32, 0, stream>>>(
      bhi, blo, xhi, xlo, b2, x2, beta, yp, part);

  // fixed-order reduction over slices + divide
  grnn_reduce<<<(BATCH * OUT_DIM + 255) / 256, 256, 0, stream>>>(part, (float*)d_out);
}
